// GRUReadout_43550968382275
// MI455X (gfx1250) — matrix operand order repaired and checked by reading, not compiled, not run on hardware
//
#include <hip/hip_runtime.h>
#include <hip/hip_bf16.h>

// Problem constants (from reference): B=64, S=512, IN=1024, H=1024, OUT=512
#define GBATCH 64
#define GS     512
#define GIN    1024
#define GH     1024
#define GOUT   512
#define GM     (GBATCH * GS)   // 32768 rows for the big GEMMs

typedef __attribute__((ext_vector_type(16))) __bf16 v16bf;
typedef __attribute__((ext_vector_type(8)))  __bf16 v8bf;
typedef __attribute__((ext_vector_type(8)))  float  v8f;
typedef __attribute__((ext_vector_type(4)))  unsigned v4u;
typedef __attribute__((ext_vector_type(8)))  unsigned v8u;

__device__ __forceinline__ v8f wmma_bf16(v16bf a, v16bf b, v8f c) {
  // D = A(16x32 bf16) * B(32x16 bf16) + C(16x16 f32)
  return __builtin_amdgcn_wmma_f32_16x16x32_bf16(false, a, false, b, (short)0, c,
                                                 false, false);
}

__device__ __forceinline__ float bf2f(__hip_bfloat16 v) { return __bfloat162float(v); }

// CDNA5 async copy: 16B/lane global -> LDS, tracked by ASYNCcnt (no VGPR data).
__device__ __forceinline__ void async_copy_b128(const __hip_bfloat16* g, __bf16* l) {
  const unsigned lds_off = (unsigned)(unsigned long long)(uintptr_t)l;  // low 32b = LDS offset
  const unsigned long long ga = (unsigned long long)(uintptr_t)g;
  asm volatile("global_load_async_to_lds_b128 %0, %1, off"
               :
               : "v"(lds_off), "v"(ga)
               : "memory");
}
__device__ __forceinline__ void wait_async0() {
  asm volatile("s_wait_asynccnt 0x0" ::: "memory");
}
__device__ __forceinline__ void wait_tensor0() {
#if __has_builtin(__builtin_amdgcn_s_wait_tensorcnt)
  __builtin_amdgcn_s_wait_tensorcnt(0);
#else
  asm volatile("s_wait_tensorcnt 0x0" ::: "memory");
#endif
}

// CDNA5 Tensor Data Mover: one scalar DMA of a 3D tile (tile_d0 contiguous
// elems x tile_d1 rows x tile_d2 planes, 2B elements) into LDS with hardware
// row padding. Descriptor per CDNA5 ISA. Per-wave op
// (EXEC ignored) -> caller must issue from exactly one wave per block.
__device__ __forceinline__ void tdm_load_3d_2B(
    unsigned lds_off, const void* gaddr,
    unsigned tensor_d0, unsigned tensor_d1, unsigned tensor_d2,
    unsigned tile_d0, unsigned tile_d1, unsigned tile_d2,
    unsigned long long stride_d0, unsigned long long stride_d1,
    unsigned pad_interval, unsigned pad_amount) {
  const unsigned long long ga = (unsigned long long)(uintptr_t)gaddr;
  v4u g0; v8u g1; v4u g2; v4u g3;
  // group0: count=1 | lds_addr | global_addr | type=2 ("image")
  g0[0] = 1u;
  g0[1] = lds_off;
  g0[2] = (unsigned)ga;
  g0[3] = (unsigned)(ga >> 32) | (2u << 30);
  // group1: data_size=2B, pad_enable, pad_interval/amount, dims/strides
  g1[0] = (1u << 16) | (1u << 20) | (pad_interval << 22) | (pad_amount << 25);
  g1[1] = (tensor_d0 & 0xFFFFu) << 16;
  g1[2] = (tensor_d0 >> 16) | ((tensor_d1 & 0xFFFFu) << 16);
  g1[3] = (tensor_d1 >> 16) | (tile_d0 << 16);
  g1[4] = tile_d1 | (tile_d2 << 16);
  g1[5] = (unsigned)stride_d0;
  g1[6] = (unsigned)(stride_d0 >> 32) | ((unsigned)(stride_d1 & 0xFFFFull) << 16);
  g1[7] = (unsigned)(stride_d1 >> 16);
  // group2/3: tensor_dim2, dims 3/4 unused
  g2[0] = tensor_d2; g2[1] = 0u; g2[2] = 0u; g2[3] = 0u;
  g3[0] = 0u; g3[1] = 0u; g3[2] = 0u; g3[3] = 0u;
  asm volatile("tensor_load_to_lds %0, %1, %2, %3"
               :
               : "s"(g0), "s"(g1), "s"(g2), "s"(g3)
               : "memory");
}

__device__ __forceinline__ float fast_sigmoid(float x) {
  return __builtin_amdgcn_rcpf(1.f + __expf(-x));
}
__device__ __forceinline__ float fast_tanh(float x) {
#if __has_builtin(__builtin_amdgcn_tanhf)
  return __builtin_amdgcn_tanhf(x);   // gfx1250 v_tanh_f32
#else
  const float cx = fminf(15.f, fmaxf(-15.f, x));
  const float e  = __expf(2.f * cx);
  return (e - 1.f) * __builtin_amdgcn_rcpf(e + 1.f);
#endif
}

// ---------------------------------------------------------------------------
// fp32 -> bf16 conversion (vectorized x4)
// ---------------------------------------------------------------------------
__global__ __launch_bounds__(256) void f32_to_bf16_kernel(
    const float* __restrict__ in, __hip_bfloat16* __restrict__ out, long n4) {
  long i = (long)blockIdx.x * blockDim.x + threadIdx.x;
  if (i >= n4) return;
  const float4 f = ((const float4*)in)[i];
  __hip_bfloat16 t[4];
  t[0] = __float2bfloat16(f.x);
  t[1] = __float2bfloat16(f.y);
  t[2] = __float2bfloat16(f.z);
  t[3] = __float2bfloat16(f.w);
  ((uint2*)out)[i] = *(const uint2*)t;
}

// Zero h ping/pong buffers (as 32-bit words) + grid-barrier counter
__global__ __launch_bounds__(256) void init_state_kernel(
    unsigned* __restrict__ hwords, long nwords, unsigned* __restrict__ ctr) {
  long i = (long)blockIdx.x * blockDim.x + threadIdx.x;
  if (i < nwords) hwords[i] = 0u;
  if (i == 0) *ctr = 0u;
}

// ---------------------------------------------------------------------------
// bf16 WMMA GEMM: C[M,N] = A[M,K] @ W[N,K]^T (+bias), A/W bf16 row-major.
// Block: 256 threads (8 waves), macro-tile 128x128, BK=32, double-buffered LDS
// filled by global_load_async_to_lds_b128. Wave (rowg,colg) computes a 32x64
// patch = 2x4 WMMA 16x16 tiles (8 v_wmma per k-step).
// ---------------------------------------------------------------------------
#define BM 128
#define BN 128
#define BK 32
#define LDP (BK + 8)   // 40 elems = 80B row stride (16B aligned, conflict-skewed)

template <bool STORE_BF16, bool ADD_BIAS>
__global__ __launch_bounds__(256) void gemm_bf16_kernel(
    const __hip_bfloat16* __restrict__ A, const __hip_bfloat16* __restrict__ W,
    const float* __restrict__ bias, void* __restrict__ Cout, int M, int N, int K) {
  __shared__ __bf16 ldsA[2][BM][LDP];
  __shared__ __bf16 ldsB[2][BN][LDP];

  const int tid  = threadIdx.x;
  const int lane = tid & 31;
  const int wave = tid >> 5;
  const int lr   = lane & 15;   // row/col within 16x16 tile
  const int hi   = lane >> 4;   // half-wave selector
  const int rowg = wave >> 1;   // 0..3 : 32-row group
  const int colg = wave & 1;    // 0..1 : 64-col group
  const long blockM = (long)blockIdx.y * BM;
  const long blockN = (long)blockIdx.x * BN;

  const v8f vzero = {0.f, 0.f, 0.f, 0.f, 0.f, 0.f, 0.f, 0.f};
  v8f acc[2][4];
#pragma unroll
  for (int i = 0; i < 2; ++i)
#pragma unroll
    for (int j = 0; j < 4; ++j) acc[i][j] = vzero;

  // async tile staging: 128x32 bf16 per matrix = 8 KB -> 32B (2x b128) / thread
  const int srow = tid >> 1;          // 0..127
  const int scol = (tid & 1) * 16;    // 0 or 16
  const __hip_bfloat16* Ag = A + (blockM + srow) * (long)K + scol;
  const __hip_bfloat16* Bg = W + (blockN + srow) * (long)K + scol;

#define STAGE(buf, k0)                                              \
  do {                                                              \
    async_copy_b128(Ag + (k0),     &ldsA[buf][srow][scol]);         \
    async_copy_b128(Ag + (k0) + 8, &ldsA[buf][srow][scol + 8]);     \
    async_copy_b128(Bg + (k0),     &ldsB[buf][srow][scol]);         \
    async_copy_b128(Bg + (k0) + 8, &ldsB[buf][srow][scol + 8]);     \
  } while (0)

  STAGE(0, 0);
  wait_async0();
  __syncthreads();

  int p = 0;
  for (int k0 = 0; k0 < K; k0 += BK) {
    if (k0 + BK < K) STAGE(p ^ 1, k0 + BK);  // overlap next-tile DMA with WMMA

    // A fragments: lane<16 -> K {0..7,16..23}, lane>=16 -> K {8..15,24..31}
    v16bf af[2];
#pragma unroll
    for (int mt = 0; mt < 2; ++mt) {
      const __bf16* pa = &ldsA[p][rowg * 32 + mt * 16 + lr][hi * 8];
      ((v8bf*)&af[mt])[0] = *(const v8bf*)pa;
      ((v8bf*)&af[mt])[1] = *(const v8bf*)(pa + 16);
    }
    // B fragments: lane<16 -> K {0..7,16..23}, lane>=16 -> K {8..15,24..31}
    v16bf bfr[4];
#pragma unroll
    for (int nt = 0; nt < 4; ++nt) {
      const __bf16* pb = &ldsB[p][colg * 64 + nt * 16 + lr][hi * 8];
      ((v8bf*)&bfr[nt])[0] = *(const v8bf*)pb;
      ((v8bf*)&bfr[nt])[1] = *(const v8bf*)(pb + 16);
    }
#pragma unroll
    for (int mt = 0; mt < 2; ++mt)
#pragma unroll
      for (int nt = 0; nt < 4; ++nt)
        acc[mt][nt] = wmma_bf16(af[mt], bfr[nt], acc[mt][nt]);

    wait_async0();     // our 4 async copies for the next buffer are done
    __syncthreads();   // everyone's copies + everyone's reads are done
    p ^= 1;
  }
#undef STAGE

  // C/D layout: VGPR v, lane -> (M = v + 8*hi, N = lr) within each 16x16 tile
#pragma unroll
  for (int mt = 0; mt < 2; ++mt) {
#pragma unroll
    for (int nt = 0; nt < 4; ++nt) {
      const long gn = blockN + colg * 64 + nt * 16 + lr;
      const float bv = ADD_BIAS ? bias[gn] : 0.f;
#pragma unroll
      for (int v = 0; v < 8; ++v) {
        const long gm  = blockM + rowg * 32 + mt * 16 + hi * 8 + v;
        const float val = acc[mt][nt][v] + bv;
        if (STORE_BF16)
          ((__hip_bfloat16*)Cout)[gm * (long)N + gn] = __float2bfloat16(val);
        else
          ((float*)Cout)[gm * (long)N + gn] = val;
      }
    }
  }
}

// ---------------------------------------------------------------------------
// Persistent GRU recurrence. Grid = 32 blocks (2 row-halves of B x 16 slices
// of H). Per step each block computes a 32x64 patch of the three gate
// pre-activations hg = h @ w_hh^T (w_hh bf16 resides in the 192MB L2 across
// all 512 steps), fuses sigmoid/tanh gates, writes h_next + hs. Cross-block
// step dependency uses a monotonic atomic-counter barrier (32 co-resident
// blocks -> deadlock-free). Weight tiles fetched by the Tensor Data Mover:
// ONE tensor_load_to_lds per k-step moves the full 3-gate tile (32 x 64 x 3,
// z-stride = H*H) into LDS with hardware padding to the 40-elem row pitch.
// Double buffered; one workgroup barrier per k-step.
// ---------------------------------------------------------------------------
#define SBK  32
#define SLDP (SBK + 8)
#define WBUF_BYTES (3 * 64 * SLDP * 2)   // 15360 B per buffer

__global__ __launch_bounds__(256) void gru_persistent_kernel(
    const __hip_bfloat16* __restrict__ xg,   // [B*S, 3H] bf16 (no bias)
    const __hip_bfloat16* __restrict__ whh,  // [3H, H] bf16
    const float* __restrict__ b_ih,          // [3H]
    const float* __restrict__ b_hh,          // [3H]
    __hip_bfloat16* __restrict__ h0,         // [B,H] ping
    __hip_bfloat16* __restrict__ h1,         // [B,H] pong
    __hip_bfloat16* __restrict__ hs,         // [B,S,H]
    unsigned* __restrict__ bar) {
  __shared__ __bf16 ldsW[2][3][64][SLDP];  // 2 bufs x 3 gates x (64 cols x 32 k)

  const int tid  = threadIdx.x;
  const int lane = tid & 31;
  const int wave = tid >> 5;
  const int lr   = lane & 15;
  const int hi   = lane >> 4;
  const int rt   = wave >> 2;                  // 0..1 : 16-row tile
  const int ct   = wave & 3;                   // 0..3 : 16-col tile
  const int jBase   = (blockIdx.x & 15) * 64;  // slice of H
  const int rowBase = (blockIdx.x >> 4) * 32;  // half of B
  const unsigned nblocks = gridDim.x;
  const bool issuer = (wave == 0);             // TDM is per-wave; issue once

  const unsigned ldsWBase = (unsigned)(unsigned long long)(uintptr_t)&ldsW[0][0][0][0];
  const __hip_bfloat16* Wtile = whh + (long)jBase * GH;  // row jBase, gate 0

#define WSTAGE(buf, k0)                                                     \
  tdm_load_3d_2B(ldsWBase + (unsigned)(buf) * WBUF_BYTES, Wtile + (k0),     \
                 /*tensor*/ GH, 64u, 3u, /*tile*/ SBK, 64u, 3u,             \
                 /*stride0*/ (unsigned long long)GH,                        \
                 /*stride1*/ (unsigned long long)GH * GH,                   \
                 /*pad: +4 DW every 16 DW*/ 3u, 3u)

  for (int t = 0; t < GS; ++t) {
    const __hip_bfloat16* hp = (t & 1) ? h1 : h0;
    __hip_bfloat16*       hn = (t & 1) ? h0 : h1;

    const v8f vzero = {0.f, 0.f, 0.f, 0.f, 0.f, 0.f, 0.f, 0.f};
    v8f acc[3];
    acc[0] = vzero; acc[1] = vzero; acc[2] = vzero;

    if (issuer) WSTAGE(0, 0);
    wait_tensor0();
    __syncthreads();

    int p = 0;
    for (int k0 = 0; k0 < GH; k0 += SBK) {
      if (issuer && (k0 + SBK < GH)) WSTAGE(p ^ 1, k0 + SBK);  // overlap DMA

      // A fragment straight from global h (128 KB, L0/L2 resident)
      v16bf af;
      const __hip_bfloat16* pa = hp + (long)(rowBase + rt * 16 + lr) * GH + k0;
      ((v8bf*)&af)[0] = *(const v8bf*)(pa + hi * 8);
      ((v8bf*)&af)[1] = *(const v8bf*)(pa + 16 + hi * 8);

#pragma unroll
      for (int g = 0; g < 3; ++g) {
        v16bf bfr;
        const __bf16* pb = &ldsW[p][g][ct * 16 + lr][hi * 8];
        ((v8bf*)&bfr)[0] = *(const v8bf*)pb;
        ((v8bf*)&bfr)[1] = *(const v8bf*)(pb + 16);
        acc[g] = wmma_bf16(af, bfr, acc[g]);
      }
      wait_tensor0();    // wave0 waits its DMA; others pass (cnt==0)
      __syncthreads();   // DMA'd buffer visible to all waves
      p ^= 1;
    }

    // Fused gates (f32 math): r,z = sigmoid; n = tanh; h' = (1-z)n + z h
    const int j = jBase + ct * 16 + lr;
#pragma unroll
    for (int v = 0; v < 8; ++v) {
      const int  b  = rowBase + rt * 16 + hi * 8 + v;
      const long m  = (long)b * GS + t;
      const long x3 = m * (long)(3 * GH);
      const float xr = bf2f(xg[x3 + j])          + b_ih[j];
      const float xz = bf2f(xg[x3 + GH + j])     + b_ih[GH + j];
      const float xn = bf2f(xg[x3 + 2 * GH + j]) + b_ih[2 * GH + j];
      const float hr = acc[0][v] + b_hh[j];
      const float hz = acc[1][v] + b_hh[GH + j];
      const float hg = acc[2][v] + b_hh[2 * GH + j];
      const float r = fast_sigmoid(xr + hr);
      const float z = fast_sigmoid(xz + hz);
      const float n = fast_tanh(xn + r * hg);
      const float hprev = bf2f(hp[(long)b * GH + j]);
      const float hnew  = (1.f - z) * n + z * hprev;
      const __hip_bfloat16 hb = __float2bfloat16(hnew);
      hn[(long)b * GH + j] = hb;
      hs[m * (long)GH + j] = hb;
    }

    // device-wide barrier (monotonic counter; 32 resident blocks)
    __syncthreads();
    if (tid == 0) {
      __threadfence();
      atomicAdd(bar, 1u);
      const unsigned target = nblocks * (unsigned)(t + 1);
      while (atomicAdd(bar, 0u) < target) {
        __builtin_amdgcn_s_sleep(2);
      }
      __threadfence();
    }
    __syncthreads();
  }
#undef WSTAGE
}

// ---------------------------------------------------------------------------
extern "C" void kernel_launch(void* const* d_in, const int* in_sizes, int n_in,
                              void* d_out, int out_size, void* d_ws, size_t ws_size,
                              hipStream_t stream) {
  (void)in_sizes; (void)n_in; (void)out_size; (void)ws_size;
  const float* x     = (const float*)d_in[0];
  const float* w_ih  = (const float*)d_in[1];
  const float* w_hh  = (const float*)d_in[2];
  const float* b_ih  = (const float*)d_in[3];
  const float* b_hh  = (const float*)d_in[4];
  const float* w_out = (const float*)d_in[5];
  const float* b_out = (const float*)d_in[6];

  char* ws = (char*)d_ws;
  size_t off = 0;
  auto carve = [&](size_t bytes) -> void* {
    void* p = ws + off;
    off += (bytes + 255) & ~(size_t)255;
    return p;
  };
  __hip_bfloat16* xb    = (__hip_bfloat16*)carve((size_t)GM * GIN * 2);
  __hip_bfloat16* wihb  = (__hip_bfloat16*)carve((size_t)3 * GH * GIN * 2);
  __hip_bfloat16* whhb  = (__hip_bfloat16*)carve((size_t)3 * GH * GH * 2);
  __hip_bfloat16* woutb = (__hip_bfloat16*)carve((size_t)GOUT * GH * 2);
  __hip_bfloat16* xgb   = (__hip_bfloat16*)carve((size_t)GM * 3 * GH * 2);
  __hip_bfloat16* h0    = (__hip_bfloat16*)carve((size_t)GBATCH * GH * 2);
  __hip_bfloat16* h1    = (__hip_bfloat16*)carve((size_t)GBATCH * GH * 2);
  __hip_bfloat16* hsb   = (__hip_bfloat16*)carve((size_t)GBATCH * GS * GH * 2);
  unsigned* bar         = (unsigned*)carve(256);

  auto cvt = [&](const float* src, __hip_bfloat16* dst, long n) {
    long n4 = n / 4;
    f32_to_bf16_kernel<<<dim3((unsigned)((n4 + 255) / 256)), dim3(256), 0, stream>>>(
        src, dst, n4);
  };
  cvt(x,     xb,    (long)GM * GIN);
  cvt(w_ih,  wihb,  (long)3 * GH * GIN);
  cvt(w_hh,  whhb,  (long)3 * GH * GH);
  cvt(w_out, woutb, (long)GOUT * GH);

  // h0 and h1 are contiguous carves; zero both + barrier counter
  const long nwords = ((long)GBATCH * GH * 2 * 2) / 4;
  init_state_kernel<<<dim3((unsigned)((nwords + 255) / 256)), dim3(256), 0, stream>>>(
      (unsigned*)h0, nwords, bar);

  // xg = x @ w_ih^T  (bias folded into step kernel), stored bf16
  gemm_bf16_kernel<true, false><<<dim3(3 * GH / BN, GM / BM), dim3(256), 0, stream>>>(
      xb, wihb, nullptr, xgb, GM, 3 * GH, GIN);

  // Sequential recurrence, persistent kernel (32 blocks)
  gru_persistent_kernel<<<dim3(32), dim3(256), 0, stream>>>(
      xgb, whhb, b_ih, b_hh, h0, h1, hsb, bar);

  // out = hs @ w_out^T + b_out, f32
  gemm_bf16_kernel<false, true><<<dim3(GOUT / BN, GM / BM), dim3(256), 0, stream>>>(
      hsb, woutb, b_out, d_out, GM, GOUT, GH);
}
